// TransformerBlock_5231270167156
// MI455X (gfx1250) — matrix operand order repaired and checked by reading, not compiled, not run on hardware
//
#include <hip/hip_runtime.h>

typedef __bf16 bf16_t;
typedef __attribute__((ext_vector_type(8)))  __bf16 v8bf;
typedef __attribute__((ext_vector_type(16))) __bf16 v16bf;
typedef __attribute__((ext_vector_type(8)))  float  v8f;

#define S_LEN 4096
#define D_DIM 1024
#define H_NUM 16
#define HS_DIM 64
#define FF_DIM 4096

#define BM 128
#define BN 128
#define BK 64
#define LDS_A 72
#define LDS_B 72
#define LDK 72

__device__ __forceinline__ v16bf cat8(v8bf lo, v8bf hi) {
  return __builtin_shufflevector(lo, hi, 0,1,2,3,4,5,6,7,8,9,10,11,12,13,14,15);
}

// ---- CDNA5 async global->LDS copy (ASYNCcnt-tracked, no VGPR data path) ----
typedef __attribute__((address_space(3))) char lds_char_t;

__device__ __forceinline__ void async_b128(bf16_t* lds_dst, const bf16_t* gsrc) {
  unsigned l = (unsigned)(uintptr_t)(lds_char_t*)(char*)lds_dst;
  asm volatile("global_load_async_to_lds_b128 %0, %1, off"
               :: "v"(l), "v"(gsrc) : "memory");
}
// wait until at most the (in-flight prefetch) 8 chunks remain: current tile landed
__device__ __forceinline__ void wait_async8() {
  asm volatile("s_wait_asynccnt 0x8" ::: "memory");
}
__device__ __forceinline__ void wait_async0() {
  asm volatile("s_wait_asynccnt 0x0" ::: "memory");
}

// ---- elementwise f32 -> bf16 ----
__global__ __launch_bounds__(256)
void cvt_bf16_kernel(const float* __restrict__ in, bf16_t* __restrict__ out)
{
  int i = (blockIdx.x * 256 + threadIdx.x) * 4;
  float4 v = *(const float4*)(in + i);
  out[i + 0] = (bf16_t)v.x; out[i + 1] = (bf16_t)v.y;
  out[i + 2] = (bf16_t)v.z; out[i + 3] = (bf16_t)v.w;
}

// ---- Wt[n][k] = (bf16)W[k][n] ----
__global__ __launch_bounds__(256)
void transpose_cvt_kernel(const float* __restrict__ W, bf16_t* __restrict__ Wt,
                          int K, int N)
{
  __shared__ float t[32][33];
  int n0 = blockIdx.x * 32;
  int k0 = blockIdx.y * 32;
  int c  = threadIdx.x & 31;
  int r0 = threadIdx.x >> 5;
  for (int i = 0; i < 4; ++i) {
    int r = r0 + i * 8;
    t[r][c] = W[(size_t)(k0 + r) * N + n0 + c];
  }
  __syncthreads();
  for (int i = 0; i < 4; ++i) {
    int r = r0 + i * 8;
    Wt[(size_t)(n0 + r) * K + k0 + c] = (bf16_t)t[c][r];
  }
}

// C = A(bf16,[M][K]) @ Wt(bf16,[N][K])^T + bias, optional residual(f32)/relu.
// OMODE: 0 = f32 [M][N], 1 = bf16 [M][N], 2 = bf16 [h][row][hs], 3 = bf16 [h][hs][row]
template<int OMODE, bool HAS_RES, bool RELU>
__global__ __launch_bounds__(256)
void gemm_wmma_kernel(const bf16_t* __restrict__ A, const bf16_t* __restrict__ Wt,
                      const float* __restrict__ bias, const float* __restrict__ res,
                      float* __restrict__ outF, bf16_t* __restrict__ outB,
                      int M, int K, int N)
{
  __shared__ bf16_t As[2][BM * LDS_A];
  __shared__ bf16_t Bs[2][BN * LDS_B];
  const int tileN = blockIdx.x * BN;
  const int tileM = blockIdx.y * BM;
  const int tid   = threadIdx.x;
  const int lane  = tid & 31;
  const int wid   = tid >> 5;
  const int waveM = wid >> 1;
  const int waveN = wid & 1;
  const int lm    = lane & 15;
  const int g     = lane >> 4;

  v8f acc[2][4];
  for (int mf = 0; mf < 2; ++mf)
    for (int nf = 0; nf < 4; ++nf)
      for (int i = 0; i < 8; ++i) acc[mf][nf][i] = 0.f;

  const int nk = K / BK;
  auto stage = [&](int kt, int buf) {
    const int k0 = kt * BK;
    for (int p = 0; p < 4; ++p) {
      int idx = p * 256 + tid;           // 0..1023
      int row = idx >> 3;                // 0..127
      int c   = (idx & 7) << 3;          // 16B chunks
      async_b128(&As[buf][row * LDS_A + c], A  + (size_t)(tileM + row) * K + k0 + c);
      async_b128(&Bs[buf][row * LDS_B + c], Wt + (size_t)(tileN + row) * K + k0 + c);
    }
  };

  stage(0, 0);                                  // prologue prefetch
  for (int kt = 0; kt < nk; ++kt) {
    const int buf = kt & 1;
    if (kt + 1 < nk) { stage(kt + 1, buf ^ 1); wait_async8(); }
    else             { wait_async0(); }
    __syncthreads();

    for (int ks = 0; ks < 2; ++ks) {
      v16bf a[2];
      for (int mf = 0; mf < 2; ++mf) {
        const bf16_t* base = &As[buf][(waveM * 32 + mf * 16 + lm) * LDS_A + ks * 32];
        a[mf] = cat8(*(const v8bf*)(base + g * 16), *(const v8bf*)(base + g * 16 + 8));
      }
      for (int nf = 0; nf < 4; ++nf) {
        const bf16_t* base = &Bs[buf][(waveN * 64 + nf * 16 + lm) * LDS_B + ks * 32 + g * 16];
        v16bf b = cat8(*(const v8bf*)(base), *(const v8bf*)(base + 8));
        for (int mf = 0; mf < 2; ++mf) {
          acc[mf][nf] = __builtin_amdgcn_wmma_f32_16x16x32_bf16(
              false, a[mf], false, b, (short)0, acc[mf][nf], false, false);
        }
      }
    }
    __syncthreads();   // all waves done reading buf before it is overwritten
  }

  for (int mf = 0; mf < 2; ++mf) {
    for (int nf = 0; nf < 4; ++nf) {
      int col = tileN + waveN * 64 + nf * 16 + lm;
      float bcol = bias[col];
      for (int r = 0; r < 8; ++r) {
        int row = tileM + waveM * 32 + mf * 16 + g * 8 + r;
        float v = acc[mf][nf][r] + bcol;
        if constexpr (HAS_RES) v += res[(size_t)row * N + col];
        if constexpr (RELU)    v = fmaxf(v, 0.f);
        if constexpr (OMODE == 0) {
          outF[(size_t)row * N + col] = v;
        } else if constexpr (OMODE == 1) {
          outB[(size_t)row * N + col] = (bf16_t)v;
        } else if constexpr (OMODE == 2) {
          int hh = col >> 6, hs = col & 63;
          outB[((size_t)hh * M + row) * HS_DIM + hs] = (bf16_t)v;
        } else {
          int hh = col >> 6, hs = col & 63;
          outB[((size_t)hh * HS_DIM + hs) * M + row] = (bf16_t)v;
        }
      }
    }
  }
}

// Flash attention, causal. q,k: bf16 [H][S][64]; v: bf16 [H][64][S]; ctx: bf16 [S][D].
__global__ __launch_bounds__(128)
void attn_wmma_kernel(const bf16_t* __restrict__ qg, const bf16_t* __restrict__ kg,
                      const bf16_t* __restrict__ vg, bf16_t* __restrict__ ctx)
{
  __shared__ bf16_t Kb[2][64 * LDK];     // [key][hs]
  __shared__ bf16_t Vt[2][64 * LDK];     // [hs][key]
  __shared__ bf16_t Ps[4 * 16 * LDK];    // per-wave P scratch

  const int h    = blockIdx.y;
  const int qblk = blockIdx.x;
  const int tid  = threadIdx.x;
  const int lane = tid & 31;
  const int wid  = tid >> 5;
  const int lm   = lane & 15;
  const int g    = lane >> 4;
  const int qrow0 = qblk * 64 + wid * 16;

  v16bf aq[2];
  {
    const bf16_t* qrow = qg + ((size_t)h * S_LEN + qrow0 + lm) * HS_DIM;
    for (int ks = 0; ks < 2; ++ks)
      aq[ks] = cat8(*(const v8bf*)(qrow + ks * 32 + g * 16),
                    *(const v8bf*)(qrow + ks * 32 + g * 16 + 8));
  }

  v8f o[4];
  for (int nf = 0; nf < 4; ++nf) for (int i = 0; i < 8; ++i) o[nf][i] = 0.f;
  float mrun[8], lrun[8];
  for (int r = 0; r < 8; ++r) { mrun[r] = -1e30f; lrun[r] = 0.f; }

  const int nkb = qblk + 1;
  auto stageKV = [&](int kbi, int buf) {
    const int kbase = kbi * 64;
    for (int p = 0; p < 4; ++p) {
      int idx = p * 128 + tid;
      int row = idx >> 3;
      int c   = (idx & 7) << 3;
      async_b128(&Kb[buf][row * LDK + c],
                 kg + ((size_t)h * S_LEN + kbase + row) * HS_DIM + c);
      async_b128(&Vt[buf][row * LDK + c],
                 vg + ((size_t)h * HS_DIM + row) * S_LEN + kbase + c);
    }
  };

  stageKV(0, 0);
  for (int kbi = 0; kbi < nkb; ++kbi) {
    const int buf = kbi & 1;
    const int kbase = kbi * 64;
    if (kbi + 1 < nkb) { stageKV(kbi + 1, buf ^ 1); wait_async8(); }
    else               { wait_async0(); }
    __syncthreads();

    v8f s[4];
    for (int nf = 0; nf < 4; ++nf) {
      for (int i = 0; i < 8; ++i) s[nf][i] = 0.f;
      for (int ks = 0; ks < 2; ++ks) {
        const bf16_t* base = &Kb[buf][(nf * 16 + lm) * LDK + ks * 32 + g * 16];
        v16bf b = cat8(*(const v8bf*)base, *(const v8bf*)(base + 8));
        s[nf] = __builtin_amdgcn_wmma_f32_16x16x32_bf16(
            false, aq[ks], false, b, (short)0, s[nf], false, false);
      }
    }
    for (int nf = 0; nf < 4; ++nf) {
      int col = kbase + nf * 16 + lm;
      for (int r = 0; r < 8; ++r) {
        int row = qrow0 + g * 8 + r;
        float v = s[nf][r] * 0.125f;
        s[nf][r] = (col > row) ? -1e30f : v;
      }
    }
    for (int r = 0; r < 8; ++r) {
      float mx = fmaxf(fmaxf(s[0][r], s[1][r]), fmaxf(s[2][r], s[3][r]));
      mx = fmaxf(mx, __shfl_xor(mx, 1, 32));
      mx = fmaxf(mx, __shfl_xor(mx, 2, 32));
      mx = fmaxf(mx, __shfl_xor(mx, 4, 32));
      mx = fmaxf(mx, __shfl_xor(mx, 8, 32));
      float mnew  = fmaxf(mrun[r], mx);
      float alpha = __expf(mrun[r] - mnew);
      float rs = 0.f;
      for (int nf = 0; nf < 4; ++nf) { float p = __expf(s[nf][r] - mnew); s[nf][r] = p; rs += p; }
      rs += __shfl_xor(rs, 1, 32); rs += __shfl_xor(rs, 2, 32);
      rs += __shfl_xor(rs, 4, 32); rs += __shfl_xor(rs, 8, 32);
      lrun[r] = lrun[r] * alpha + rs;
      mrun[r] = mnew;
      for (int nf = 0; nf < 4; ++nf) o[nf][r] *= alpha;
    }
    bf16_t* ps = &Ps[wid * 16 * LDK];
    for (int nf = 0; nf < 4; ++nf)
      for (int r = 0; r < 8; ++r)
        ps[(g * 8 + r) * LDK + nf * 16 + lm] = (bf16_t)s[nf][r];
    v16bf ap[2];
    for (int ks = 0; ks < 2; ++ks) {
      const bf16_t* base = &ps[lm * LDK + ks * 32];
      ap[ks] = cat8(*(const v8bf*)(base + g * 16), *(const v8bf*)(base + g * 16 + 8));
    }
    for (int nf = 0; nf < 4; ++nf)
      for (int ks = 0; ks < 2; ++ks) {
        const bf16_t* base = &Vt[buf][(nf * 16 + lm) * LDK + ks * 32 + g * 16];
        v16bf b = cat8(*(const v8bf*)base, *(const v8bf*)(base + 8));
        o[nf] = __builtin_amdgcn_wmma_f32_16x16x32_bf16(
            false, ap[ks], false, b, (short)0, o[nf], false, false);
      }
    __syncthreads();   // done reading buf before it is overwritten next+1 iter
  }

  for (int nf = 0; nf < 4; ++nf)
    for (int r = 0; r < 8; ++r) {
      int row = qrow0 + g * 8 + r;
      ctx[(size_t)row * D_DIM + h * HS_DIM + nf * 16 + lm] = (bf16_t)(o[nf][r] / lrun[r]);
    }
}

// LayerNorm; writes f32 (and optional bf16 copy for downstream WMMA GEMMs).
__global__ __launch_bounds__(256)
void layernorm_kernel(const float* __restrict__ in, const float* __restrict__ gam,
                      const float* __restrict__ bet, float* __restrict__ outF,
                      bf16_t* __restrict__ outB)
{
  __shared__ float sbuf[256], s2buf[256];
  const int row = blockIdx.x;
  const int t = threadIdx.x;
  const float* xr = in + (size_t)row * D_DIM;
  float v[4]; float s = 0.f, ss = 0.f;
  for (int i = 0; i < 4; ++i) { float x = xr[t + i * 256]; v[i] = x; s += x; ss += x * x; }
  sbuf[t] = s; s2buf[t] = ss;
  __syncthreads();
  for (int stride = 128; stride > 0; stride >>= 1) {
    if (t < stride) { sbuf[t] += sbuf[t + stride]; s2buf[t] += s2buf[t + stride]; }
    __syncthreads();
  }
  float mu  = sbuf[0] * (1.0f / D_DIM);
  float var = s2buf[0] * (1.0f / D_DIM) - mu * mu;
  float inv = rsqrtf(var + 1e-5f);
  for (int i = 0; i < 4; ++i) {
    int c = t + i * 256;
    float y = (v[i] - mu) * inv * gam[c] + bet[c];
    outF[(size_t)row * D_DIM + c] = y;
    if (outB) outB[(size_t)row * D_DIM + c] = (bf16_t)y;
  }
}

extern "C" void kernel_launch(void* const* d_in, const int* in_sizes, int n_in,
                              void* d_out, int out_size, void* d_ws, size_t ws_size,
                              hipStream_t stream)
{
  (void)in_sizes; (void)n_in; (void)out_size; (void)ws_size;
  const float* x   = (const float*)d_in[0];
  const float* wq  = (const float*)d_in[1];
  const float* bq  = (const float*)d_in[2];
  const float* wk  = (const float*)d_in[3];
  const float* bk  = (const float*)d_in[4];
  const float* wv  = (const float*)d_in[5];
  const float* bv  = (const float*)d_in[6];
  const float* wo  = (const float*)d_in[7];
  const float* bo  = (const float*)d_in[8];
  const float* g1  = (const float*)d_in[9];
  const float* b1  = (const float*)d_in[10];
  const float* wf1 = (const float*)d_in[11];
  const float* bf1 = (const float*)d_in[12];
  const float* wf2 = (const float*)d_in[13];
  const float* bf2 = (const float*)d_in[14];
  const float* g2  = (const float*)d_in[15];
  const float* b2  = (const float*)d_in[16];

  char* ws = (char*)d_ws;
  const size_t MB = 1024u * 1024u;
  bf16_t* xb   = (bf16_t*)(ws + 0);        // [S][D] bf16, 8 MB
  bf16_t* wqT  = (bf16_t*)(ws + 8  * MB);  // [D][D] bf16, 2 MB each
  bf16_t* wkT  = (bf16_t*)(ws + 10 * MB);
  bf16_t* wvT  = (bf16_t*)(ws + 12 * MB);
  bf16_t* woT  = (bf16_t*)(ws + 14 * MB);
  bf16_t* wf1T = (bf16_t*)(ws + 16 * MB);  // [FF][D] bf16, 8 MB
  bf16_t* wf2T = (bf16_t*)(ws + 24 * MB);  // [D][FF] bf16, 8 MB
  bf16_t* qb   = (bf16_t*)(ws + 32 * MB);  // [H][S][64]
  bf16_t* kb   = (bf16_t*)(ws + 40 * MB);  // [H][S][64]
  bf16_t* vt   = (bf16_t*)(ws + 48 * MB);  // [H][64][S]
  bf16_t* ctxb = (bf16_t*)(ws + 56 * MB);  // [S][D] bf16
  float*  y1   = (float*)(ws + 64 * MB);   // [S][D] f32
  float*  h1   = (float*)(ws + 80 * MB);   // [S][D] f32
  bf16_t* h1b  = (bf16_t*)(ws + 96 * MB);  // [S][D] bf16
  bf16_t* ffb  = (bf16_t*)(ws + 104 * MB); // [S][FF] bf16, 32 MB
  float*  y2   = (float*)(ws + 136 * MB);  // [S][D] f32

  dim3 blk(256);
  dim3 gD(D_DIM / BN, S_LEN / BM);    // (8, 32)
  dim3 gF(FF_DIM / BN, S_LEN / BM);   // (32, 32)

  cvt_bf16_kernel<<<(S_LEN * D_DIM) / 1024, blk, 0, stream>>>(x, xb);
  transpose_cvt_kernel<<<dim3(D_DIM / 32, D_DIM / 32), blk, 0, stream>>>(wq, wqT, D_DIM, D_DIM);
  transpose_cvt_kernel<<<dim3(D_DIM / 32, D_DIM / 32), blk, 0, stream>>>(wk, wkT, D_DIM, D_DIM);
  transpose_cvt_kernel<<<dim3(D_DIM / 32, D_DIM / 32), blk, 0, stream>>>(wv, wvT, D_DIM, D_DIM);
  transpose_cvt_kernel<<<dim3(D_DIM / 32, D_DIM / 32), blk, 0, stream>>>(wo, woT, D_DIM, D_DIM);
  transpose_cvt_kernel<<<dim3(FF_DIM / 32, D_DIM / 32), blk, 0, stream>>>(wf1, wf1T, D_DIM, FF_DIM);
  transpose_cvt_kernel<<<dim3(D_DIM / 32, FF_DIM / 32), blk, 0, stream>>>(wf2, wf2T, FF_DIM, D_DIM);

  gemm_wmma_kernel<2, false, false><<<gD, blk, 0, stream>>>(xb, wqT, bq, nullptr, nullptr, qb,
                                                            S_LEN, D_DIM, D_DIM);
  gemm_wmma_kernel<2, false, false><<<gD, blk, 0, stream>>>(xb, wkT, bk, nullptr, nullptr, kb,
                                                            S_LEN, D_DIM, D_DIM);
  gemm_wmma_kernel<3, false, false><<<gD, blk, 0, stream>>>(xb, wvT, bv, nullptr, nullptr, vt,
                                                            S_LEN, D_DIM, D_DIM);

  attn_wmma_kernel<<<dim3(S_LEN / 64, H_NUM), dim3(128), 0, stream>>>(qb, kb, vt, ctxb);

  gemm_wmma_kernel<0, true, false><<<gD, blk, 0, stream>>>(ctxb, woT, bo, x, y1, nullptr,
                                                           S_LEN, D_DIM, D_DIM);
  layernorm_kernel<<<S_LEN, blk, 0, stream>>>(y1, g1, b1, h1, h1b);

  gemm_wmma_kernel<1, false, true><<<gF, blk, 0, stream>>>(h1b, wf1T, bf1, nullptr, nullptr, ffb,
                                                           S_LEN, D_DIM, FF_DIM);
  gemm_wmma_kernel<0, true, false><<<gD, blk, 0, stream>>>(ffb, wf2T, bf2, h1, y2, nullptr,
                                                           S_LEN, FF_DIM, D_DIM);
  layernorm_kernel<<<S_LEN, blk, 0, stream>>>(y2, g2, b2, (float*)d_out, nullptr);
}
